// ListMapper_26414048871089
// MI455X (gfx1250) — matrix operand order repaired and checked by reading, not compiled, not run on hardware
//
#include <hip/hip_runtime.h>
#include <math.h>

// ---------------------------------------------------------------------------
// Ragged RNN (ListMapper):  B=32, T=1024, D_IN=D_STATE=D_OUT=1024
//   Phase 1: xh = x @ Wxh + bh                (big GEMM, bf16 WMMA)
//   Phase 2: for t: h = tanh(xh_t + s @ Whh)  (1024 serial small GEMM steps)
//   Phase 3: out = mask(H @ Who + bo)         (big GEMM, bf16 WMMA)
// Workspace (~198MB): xh f32 128MB | H bf16 64MB | s0/s1 f32 256KB | 3x WT bf16 6MB
// ---------------------------------------------------------------------------

typedef __attribute__((ext_vector_type(16))) __bf16 v16bf;
typedef __attribute__((ext_vector_type(8)))  float  v8f;
typedef __attribute__((ext_vector_type(4)))  int    v4i;

union ABfrag { v16bf v; uint4 u[2]; };

// --- CDNA5 async global->LDS path (guarded; falls back to ld/st staging) ----
#if defined(__has_builtin)
#if __has_builtin(__builtin_amdgcn_global_load_async_to_lds_b128)
#define USE_ASYNC_LDS 1
#endif
#endif
#ifndef USE_ASYNC_LDS
#define USE_ASYNC_LDS 0
#endif

#if USE_ASYNC_LDS
typedef __attribute__((address_space(1))) v4i GV4;   // global int4*
typedef __attribute__((address_space(3))) v4i LV4;   // LDS int4*
__device__ __forceinline__ void async_cp16(const void* g, void* l) {
  __builtin_amdgcn_global_load_async_to_lds_b128((GV4*)(void*)g, (LV4*)l, 0, 0);
}
__device__ __forceinline__ void async_wait() {
#if __has_builtin(__builtin_amdgcn_s_wait_asynccnt)
  __builtin_amdgcn_s_wait_asynccnt(0);
#else
  asm volatile("s_wait_asynccnt 0" ::: "memory");
#endif
}
#endif

__device__ __forceinline__ unsigned short f2bf(float f) {
  unsigned int u = __float_as_uint(f);
  return (unsigned short)((u + 0x7FFFu + ((u >> 16) & 1u)) >> 16);  // RNE
}
__device__ __forceinline__ unsigned pk2bf(float a, float b) {
  return (unsigned)f2bf(a) | ((unsigned)f2bf(b) << 16);
}

// ---------------------------------------------------------------------------
// Transpose + f32->bf16 convert:  WT[n][k] = bf16(W[k][n]),  W is 1024x1024.
// ---------------------------------------------------------------------------
__global__ __launch_bounds__(256)
void tconv_kernel(const float* __restrict__ W, unsigned short* __restrict__ WT) {
  __shared__ unsigned short tile[32][33];
  const int bi = blockIdx.x & 31;   // k tile
  const int bj = blockIdx.x >> 5;   // n tile
  const int e = threadIdx.x * 4;
  const int r = e >> 5, c = e & 31;
  const float4 f = *(const float4*)&W[(size_t)(bi * 32 + r) * 1024 + bj * 32 + c];
  tile[r][c + 0] = f2bf(f.x); tile[r][c + 1] = f2bf(f.y);
  tile[r][c + 2] = f2bf(f.z); tile[r][c + 3] = f2bf(f.w);
  __syncthreads();
  uint2 d;
  d.x = (unsigned)tile[c + 0][r] | ((unsigned)tile[c + 1][r] << 16);
  d.y = (unsigned)tile[c + 2][r] | ((unsigned)tile[c + 3][r] << 16);
  *(uint2*)&WT[(size_t)(bj * 32 + r) * 1024 + bi * 32 + c] = d;
}

__global__ void zero_kernel(float* __restrict__ p, int n) {
  const int i = blockIdx.x * blockDim.x + threadIdx.x;
  if (i < n) p[i] = 0.0f;
}

// ---------------------------------------------------------------------------
// Shared per-wave compute for one K-chunk of 64 (8 WMMAs: 2 K-subchunks x 4 N).
// ldsA[32][72], ldsB[256][72]: 144B row stride -> 16B-aligned, conflict-free.
// ---------------------------------------------------------------------------
#define LDSA_W 72
#define LDSB_W 72

__device__ __forceinline__ void wave_mma64(
    const unsigned short (*ldsA)[LDSA_W], const unsigned short (*ldsB)[LDSB_W],
    int wm, int wn, int half, int r, v8f acc[4]) {
#pragma unroll
  for (int ks = 0; ks < 2; ++ks) {
    const int kb = ks * 32;
    ABfrag aF;  // A 16x32 layout: lane half h, row r: K = kb + {8h+0..7, 16+8h+0..7}
    aF.u[0] = *(const uint4*)&ldsA[wm * 16 + r][kb + half * 8];
    aF.u[1] = *(const uint4*)&ldsA[wm * 16 + r][kb + 16 + half * 8];
#pragma unroll
    for (int nt = 0; nt < 4; ++nt) {
      ABfrag bF;  // B 32x16 layout: lane half h, col r: K = kb + {8h+0..7, 16+8h+0..7}
      const int bn = wn * 64 + nt * 16 + r;
      bF.u[0] = *(const uint4*)&ldsB[bn][kb + half * 8];
      bF.u[1] = *(const uint4*)&ldsB[bn][kb + 16 + half * 8];
      acc[nt] = __builtin_amdgcn_wmma_f32_16x16x32_bf16(
          false, aF.v, false, bF.v, (short)0, acc[nt], false, false);
    }
  }
}

// Stage a 256x64 bf16 B-panel row (one row of 128B per thread).
__device__ __forceinline__ void stage_B(const unsigned short* __restrict__ src,
                                        unsigned short* __restrict__ dstRow) {
#if USE_ASYNC_LDS
#pragma unroll
  for (int i = 0; i < 8; ++i) async_cp16(src + i * 8, dstRow + i * 8);
#else
  const uint4* s4 = (const uint4*)src;
  uint4* d4 = (uint4*)dstRow;
#pragma unroll
  for (int i = 0; i < 8; ++i) d4[i] = s4[i];
#endif
}

// ---------------------------------------------------------------------------
// Generic bf16-WMMA GEMM:  C[M,N] = A[M,K] * BT[N,K]^T + bias   (f32 out)
// 256 threads = 8 waves (2M x 4N); wave tile 16x64; block tile 32x256; Kc=64.
// ---------------------------------------------------------------------------
template <bool A_BF16, bool MASKED>
__global__ __launch_bounds__(256)
void gemm_wmma_kernel(const void* __restrict__ Aptr,
                      const unsigned short* __restrict__ BT,   // [N][K] bf16
                      const float* __restrict__ bias,          // [N]
                      float* __restrict__ C,
                      const int* __restrict__ row_lengths,
                      int M, int N, int K, int nblkN) {
  __shared__ unsigned short ldsA[32][LDSA_W];
  __shared__ unsigned short ldsB[256][LDSB_W];

  const int mblk = blockIdx.x / nblkN;
  const int nblk = blockIdx.x % nblkN;
  const int tid  = threadIdx.x;
  const int lane = tid & 31, wave = tid >> 5;
  const int half = lane >> 4, r = lane & 15;
  const int wm = wave >> 2, wn = wave & 3;
  const int em = (tid * 8) >> 6, ek = (tid * 8) & 63;   // A staging: 8 elems/thread

  v8f acc[4] = {};

  for (int k0 = 0; k0 < K; k0 += 64) {
    __syncthreads();
    // ---- stage A chunk 32x64 ----
    if (A_BF16) {
      const unsigned short* A = (const unsigned short*)Aptr;
      const unsigned short* src = &A[(size_t)(mblk * 32 + em) * K + k0 + ek];
#if USE_ASYNC_LDS
      async_cp16(src, &ldsA[em][ek]);
#else
      *(uint4*)&ldsA[em][ek] = *(const uint4*)src;
#endif
    } else {
      const float* A = (const float*)Aptr;
      const float4* Ap = (const float4*)&A[(size_t)(mblk * 32 + em) * K + k0 + ek];
      const float4 f0 = Ap[0], f1 = Ap[1];
      uint4 d;
      d.x = pk2bf(f0.x, f0.y); d.y = pk2bf(f0.z, f0.w);
      d.z = pk2bf(f1.x, f1.y); d.w = pk2bf(f1.z, f1.w);
      *(uint4*)&ldsA[em][ek] = d;
    }
    // ---- stage B chunk 256x64 from transposed weights (contiguous in K) ----
    {
      const unsigned short* src = &BT[(size_t)(nblk * 256 + tid) * K + k0];
      stage_B(src, &ldsB[tid][0]);
#if !USE_ASYNC_LDS
      if (k0 + 64 < K)
        __builtin_prefetch((const char*)src + 128, 0, 3);  // global_prefetch_b8
#endif
    }
#if USE_ASYNC_LDS
    async_wait();
#endif
    __syncthreads();

    wave_mma64(ldsA, ldsB, wm, wn, half, r, acc);
  }

  // ---- epilogue: bias, optional ragged mask, f32 store ----
#pragma unroll
  for (int nt = 0; nt < 4; ++nt) {
    const int col = nblk * 256 + wn * 64 + nt * 16 + r;
    const float bv = bias[col];
#pragma unroll
    for (int j = 0; j < 8; ++j) {
      const int row = mblk * 32 + wm * 16 + 8 * half + j;
      float v = acc[nt][j] + bv;
      if (MASKED) {
        const int b = row >> 10, t = row & 1023;
        if (t >= row_lengths[b]) v = 0.0f;
      }
      C[(size_t)row * N + col] = v;
    }
  }
}

// ---------------------------------------------------------------------------
// One recurrence step:  h = tanh(xh[:,t,:] + s_in @ Whh)
//   H[b][t][:] = bf16(h);  s_out = active ? h : s_in   (active: t < len[b])
// Grid: 4 blocks x 256 threads.  acc initialized from xh (already + bh).
// ---------------------------------------------------------------------------
__global__ __launch_bounds__(256)
void rnn_step_kernel(const float* __restrict__ xh,            // [B,T,D] f32
                     const float* __restrict__ s_in,          // [B,D]  f32
                     float* __restrict__ s_out,               // [B,D]  f32
                     const unsigned short* __restrict__ WhhT, // [D][D] bf16
                     unsigned short* __restrict__ H,          // [B,T,D] bf16
                     const int* __restrict__ row_lengths,
                     int t) {
  __shared__ unsigned short ldsA[32][LDSA_W];
  __shared__ unsigned short ldsB[256][LDSB_W];

  const int nblk = blockIdx.x;          // 0..3
  const int tid  = threadIdx.x;
  const int lane = tid & 31, wave = tid >> 5;
  const int half = lane >> 4, r = lane & 15;
  const int wm = wave >> 2, wn = wave & 3;
  const int em = (tid * 8) >> 6, ek = (tid * 8) & 63;

  // acc <- xh_t fragment (C-layout: VGPR j -> M = j + 8*half, N = r)
  v8f acc[4];
#pragma unroll
  for (int nt = 0; nt < 4; ++nt) {
    const int col = nblk * 256 + wn * 64 + nt * 16 + r;
#pragma unroll
    for (int j = 0; j < 8; ++j) {
      const int b = wm * 16 + 8 * half + j;                   // batch row 0..31
      acc[nt][j] = xh[((size_t)b * 1024 + t) * 1024 + col];
    }
  }

  for (int k0 = 0; k0 < 1024; k0 += 64) {
    __syncthreads();
    {  // stage s_in chunk (f32 -> bf16)
      const float4* Sp = (const float4*)&s_in[(size_t)em * 1024 + k0 + ek];
      const float4 f0 = Sp[0], f1 = Sp[1];
      uint4 d;
      d.x = pk2bf(f0.x, f0.y); d.y = pk2bf(f0.z, f0.w);
      d.z = pk2bf(f1.x, f1.y); d.w = pk2bf(f1.z, f1.w);
      *(uint4*)&ldsA[em][ek] = d;
    }
    {  // stage WhhT chunk
      const unsigned short* src = &WhhT[(size_t)(nblk * 256 + tid) * 1024 + k0];
      stage_B(src, &ldsB[tid][0]);
#if !USE_ASYNC_LDS
      if (k0 + 64 < 1024) __builtin_prefetch((const char*)src + 128, 0, 3);
#endif
    }
#if USE_ASYNC_LDS
    async_wait();
#endif
    __syncthreads();

    wave_mma64(ldsA, ldsB, wm, wn, half, r, acc);
  }

  // tanh + ragged state update + bf16 H store
#pragma unroll
  for (int nt = 0; nt < 4; ++nt) {
    const int col = nblk * 256 + wn * 64 + nt * 16 + r;
#pragma unroll
    for (int j = 0; j < 8; ++j) {
      const int b = wm * 16 + 8 * half + j;
      const float h = tanhf(acc[nt][j]);
      const bool act = (t < row_lengths[b]);
      H[((size_t)b * 1024 + t) * 1024 + col] = f2bf(h);
      const int si = b * 1024 + col;
      s_out[si] = act ? h : s_in[si];
    }
  }
}

// ---------------------------------------------------------------------------
extern "C" void kernel_launch(void* const* d_in, const int* in_sizes, int n_in,
                              void* d_out, int out_size, void* d_ws, size_t ws_size,
                              hipStream_t stream) {
  const float* x   = (const float*)d_in[0];
  const float* Wxh = (const float*)d_in[1];
  const float* Whh = (const float*)d_in[2];
  const float* bh  = (const float*)d_in[3];
  const float* Who = (const float*)d_in[4];
  const float* bo  = (const float*)d_in[5];
  const int*   rl  = (const int*)d_in[6];
  float* out = (float*)d_out;

  char* ws = (char*)d_ws;
  float*          xh   = (float*)(ws);                                   // 128MB
  unsigned short* H    = (unsigned short*)(ws + 134217728);              //  64MB
  float*          s0   = (float*)(ws + 134217728 + 67108864);            // 128KB
  float*          s1   = s0 + 32768;                                     // 128KB
  unsigned short* WxhT = (unsigned short*)(ws + 134217728 + 67108864 + 262144);
  unsigned short* WhhT = WxhT + 1048576;
  unsigned short* WhoT = WhhT + 1048576;

  // Weight transpose+convert (one-time per call) and zero initial state.
  tconv_kernel<<<1024, 256, 0, stream>>>(Wxh, WxhT);
  tconv_kernel<<<1024, 256, 0, stream>>>(Whh, WhhT);
  tconv_kernel<<<1024, 256, 0, stream>>>(Who, WhoT);
  zero_kernel<<<128, 256, 0, stream>>>(s0, 32768);

  // Phase 1: xh = x @ Wxh + bh     (M=32768, N=1024, K=1024)
  gemm_wmma_kernel<false, false><<<4096, 256, 0, stream>>>(
      (const void*)x, WxhT, bh, xh, nullptr, 32768, 1024, 1024, 4);

  // Phase 2: 1024 serial recurrence steps (ping-pong state buffers).
  for (int t = 0; t < 1024; ++t) {
    const float* sin  = (t & 1) ? s1 : s0;
    float*       sout = (t & 1) ? s0 : s1;
    rnn_step_kernel<<<4, 256, 0, stream>>>(xh, sin, sout, WhhT, H, rl, t);
  }

  // Phase 3: out = mask(H @ Who + bo)   — writes every element (zeros padding).
  gemm_wmma_kernel<true, true><<<4096, 256, 0, stream>>>(
      (const void*)H, WhoT, bo, out, rl, 32768, 1024, 1024, 4);
}
